// SorscherRNN_41738492182538
// MI455X (gfx1250) — matrix operand order repaired and checked by reading, not compiled, not run on hardware
//
#include <hip/hip_runtime.h>
#include <hip/hip_bf16.h>
#include <stdint.h>

// ---------------- problem constants (SorscherRNN) ----------------
#define NGc 4096
#define NPc 512
#define Bc  64
#define Tc  100
#define ALPHA_KW 0.1f
#define BETA_KW  0.9f

typedef __attribute__((ext_vector_type(16))) __bf16 v16bf;
typedef __attribute__((ext_vector_type(8)))  __bf16 v8bf;
typedef __attribute__((ext_vector_type(8)))  float  v8f;

// ---------------------------------------------------------------
// 1) one-shot weight downconvert fp32 -> bf16 (W_hh then W_dec)
// ---------------------------------------------------------------
__global__ void __launch_bounds__(256)
cvt_weights_kernel(const float* __restrict__ Whh, const float* __restrict__ Wdec,
                   __bf16* __restrict__ Whh_b, __bf16* __restrict__ Wdec_b)
{
    const size_t nWhh = (size_t)NGc * NGc;
    const size_t nTot = nWhh + (size_t)NPc * NGc;
    for (size_t i = (size_t)blockIdx.x * blockDim.x + threadIdx.x; i < nTot;
         i += (size_t)gridDim.x * blockDim.x) {
        if (i < nWhh) Whh_b[i]         = (__bf16)Whh[i];
        else          Wdec_b[i - nWhh] = (__bf16)Wdec[i - nWhh];
    }
}

// ---------------------------------------------------------------
// 2) h0 = p0 @ W_init.T  (tiny: 0.27 GFLOP, plain fp32), zero state
// ---------------------------------------------------------------
__global__ void __launch_bounds__(256)
init_kernel(const float* __restrict__ p0, const float* __restrict__ Winit,
            __bf16* __restrict__ h0, float* __restrict__ va, float* __restrict__ zp)
{
    const int idx = blockIdx.x * blockDim.x + threadIdx.x;   // over B*NG
    const int b = idx >> 12;            // / NG
    const int n = idx & (NGc - 1);      // % NG
    const float* pr = p0 + b * NPc;
    const float* wr = Winit + (size_t)n * NPc;
    float s = 0.f;
    #pragma unroll 4
    for (int j = 0; j < NPc; ++j) s += pr[j] * wr[j];
    h0[idx] = (__bf16)s;
    va[idx] = 0.f;
    zp[idx] = 0.f;
}

// ---------------------------------------------------------------
// 3) one RNN timestep. One wave per WG; wave owns a 64x64 C tile
//    (4 M-tiles x 4 N-tiles). Per K=32 chunk: 4 A-frag + 4 B-frag
//    loads feed 16 v_wmma_f32_16x16x32_bf16 (1:1 load:wmma).
//    grid: NG/64 = 64 blocks x 32 threads.
// ---------------------------------------------------------------
__global__ void __launch_bounds__(32)
step_kernel(const __bf16* __restrict__ h_in,   // [B, NG] bf16
            __bf16*       __restrict__ h_out,  // [B, NG] bf16
            const __bf16* __restrict__ Whh,    // [NG, NG] bf16 (row n = output cell n)
            const float*  __restrict__ Wih,    // [NG, 2]
            const float*  __restrict__ v,      // [B, T, 2]
            float*        __restrict__ va,     // [B, NG]
            float*        __restrict__ zp,     // [B, NG]
            __bf16*       __restrict__ g,      // [B, T, NG] bf16
            int t)
{
    const int lane = threadIdx.x & 31;
    const int col  = lane & 15;     // N within B tile / M row within A tile
    const int half = lane >> 4;     // lane half selects K sub-block
    const int n0   = blockIdx.x * 64;

    // B operand: B[k][n] = Whh[n0+nt*16+n][k0+k] -> contiguous 16 bf16 per lane
    const __bf16* bbase = Whh + (size_t)(n0 + col) * NGc + half * 16;
    // A operand rows m = mt*16 + col ; K = k0 + {0..15} (+16 for upper half)
    const __bf16* abase = h_in + (size_t)col * NGc + half * 16;

    v8f acc[4][4];
    #pragma unroll
    for (int mt = 0; mt < 4; ++mt)
        #pragma unroll
        for (int nt = 0; nt < 4; ++nt) acc[mt][nt] = v8f{};

    for (int k0 = 0; k0 < NGc; k0 += 32) {
        v16bf bfrag[4];
        union { v16bf v; v8bf h[2]; } a[4];
        #pragma unroll
        for (int nt = 0; nt < 4; ++nt)
            bfrag[nt] = *(const v16bf*)(bbase + (size_t)nt * 16 * NGc + k0);
        #pragma unroll
        for (int mt = 0; mt < 4; ++mt) {
            const __bf16* ap = abase + (size_t)mt * 16 * NGc + k0;
            a[mt].h[0] = *(const v8bf*)(ap);
            a[mt].h[1] = *(const v8bf*)(ap + 8);
        }
        #pragma unroll
        for (int mt = 0; mt < 4; ++mt)
            #pragma unroll
            for (int nt = 0; nt < 4; ++nt)
                acc[mt][nt] = __builtin_amdgcn_wmma_f32_16x16x32_bf16(
                    false, a[mt].v, false, bfrag[nt], (short)0, acc[mt][nt], false, false);
    }

    // fused adaptation epilogue: C elem (r,lane) -> b = mt*16 + r + 8*half,
    //                                              n = n0 + nt*16 + col
    #pragma unroll
    for (int nt = 0; nt < 4; ++nt) {
        const int n = n0 + nt * 16 + col;
        const float wi0 = Wih[n * 2 + 0];
        const float wi1 = Wih[n * 2 + 1];
        #pragma unroll
        for (int mt = 0; mt < 4; ++mt) {
            #pragma unroll
            for (int r = 0; r < 8; ++r) {
                const int b = mt * 16 + r + half * 8;
                const size_t idx = (size_t)b * NGc + n;
                const float v0 = v[(b * Tc + t) * 2 + 0];
                const float v1 = v[(b * Tc + t) * 2 + 1];
                float z = acc[mt][nt][r] + v0 * wi0 + v1 * wi1;
                const float va_old = va[idx];
                const float zp_old = zp[idx];
                z -= ALPHA_KW * va_old;                         // uses OLD va
                va[idx] = va_old + BETA_KW * (zp_old - va_old); // uses PREVIOUS z
                zp[idx] = z;
                const float s = z > 0.f ? z : 0.f;
                const __bf16 sb = (__bf16)s;
                h_out[idx] = sb;
                g[((size_t)b * Tc + t) * NGc + n] = sb;
            }
        }
    }
}

// ---------------------------------------------------------------
// 4) decoder: out = g @ W_dec.T   ([6400,4096] x [4096,512])
//    Same 64x64-per-wave tiling. grid: (6400/64, 512/64) x 32 thr.
// ---------------------------------------------------------------
__global__ void __launch_bounds__(32)
decoder_kernel(const __bf16* __restrict__ g,   // [B*T, NG] bf16 row-major
               const __bf16* __restrict__ Wd,  // [NP, NG] bf16
               float* __restrict__ out)        // [B*T, NP] fp32
{
    const int lane = threadIdx.x & 31;
    const int col  = lane & 15;
    const int half = lane >> 4;
    const int m0   = blockIdx.x * 64;
    const int n0   = blockIdx.y * 64;

    const __bf16* bbase = Wd + (size_t)(n0 + col) * NGc + half * 16;
    const __bf16* abase = g  + (size_t)(m0 + col) * NGc + half * 16;

    v8f acc[4][4];
    #pragma unroll
    for (int mt = 0; mt < 4; ++mt)
        #pragma unroll
        for (int nt = 0; nt < 4; ++nt) acc[mt][nt] = v8f{};

    for (int k0 = 0; k0 < NGc; k0 += 32) {
        v16bf bfrag[4];
        union { v16bf v; v8bf h[2]; } a[4];
        #pragma unroll
        for (int nt = 0; nt < 4; ++nt)
            bfrag[nt] = *(const v16bf*)(bbase + (size_t)nt * 16 * NGc + k0);
        #pragma unroll
        for (int mt = 0; mt < 4; ++mt) {
            const __bf16* ap = abase + (size_t)mt * 16 * NGc + k0;
            a[mt].h[0] = *(const v8bf*)(ap);
            a[mt].h[1] = *(const v8bf*)(ap + 8);
        }
        #pragma unroll
        for (int mt = 0; mt < 4; ++mt)
            #pragma unroll
            for (int nt = 0; nt < 4; ++nt)
                acc[mt][nt] = __builtin_amdgcn_wmma_f32_16x16x32_bf16(
                    false, a[mt].v, false, bfrag[nt], (short)0, acc[mt][nt], false, false);
    }

    #pragma unroll
    for (int nt = 0; nt < 4; ++nt) {
        const int p = n0 + nt * 16 + col;
        #pragma unroll
        for (int mt = 0; mt < 4; ++mt) {
            #pragma unroll
            for (int r = 0; r < 8; ++r) {
                const int m = m0 + mt * 16 + r + half * 8;
                out[(size_t)m * NPc + p] = acc[mt][nt][r];
            }
        }
    }
}

// ---------------------------------------------------------------
// host launcher
// ---------------------------------------------------------------
extern "C" void kernel_launch(void* const* d_in, const int* in_sizes, int n_in,
                              void* d_out, int out_size, void* d_ws, size_t ws_size,
                              hipStream_t stream)
{
    (void)in_sizes; (void)n_in; (void)out_size; (void)ws_size;

    const float* v      = (const float*)d_in[0];  // [B, T, 2]
    const float* p0     = (const float*)d_in[1];  // [B, NP]
    const float* W_init = (const float*)d_in[2];  // [NG, NP]
    const float* W_ih   = (const float*)d_in[3];  // [NG, 2]
    const float* W_hh   = (const float*)d_in[4];  // [NG, NG]
    const float* W_dec  = (const float*)d_in[5];  // [NP, NG]
    float* out = (float*)d_out;                   // [B, T, NP]

    // workspace partition (256B aligned)
    char* ws = (char*)d_ws;
    size_t off = 0;
    auto take = [&](size_t bytes) { char* p = ws + off; off = (off + bytes + 255) & ~(size_t)255; return p; };
    __bf16* Whh_b  = (__bf16*)take((size_t)NGc * NGc * sizeof(__bf16)); // 33.6 MB
    __bf16* Wdec_b = (__bf16*)take((size_t)NPc * NGc * sizeof(__bf16)); //  4.2 MB
    __bf16* hbuf0  = (__bf16*)take((size_t)Bc * NGc * sizeof(__bf16));
    __bf16* hbuf1  = (__bf16*)take((size_t)Bc * NGc * sizeof(__bf16));
    float*  va     = (float*) take((size_t)Bc * NGc * sizeof(float));
    float*  zp     = (float*) take((size_t)Bc * NGc * sizeof(float));
    __bf16* gbuf   = (__bf16*)take((size_t)Bc * Tc * NGc * sizeof(__bf16)); // 52.4 MB

    // 1) weights -> bf16 (re-done every call: deterministic, no caching)
    {
        const size_t nTot = (size_t)NGc * NGc + (size_t)NPc * NGc;
        const int blocks = (int)((nTot + 256 * 8 - 1) / (256 * 8)); // grid-stride
        hipLaunchKernelGGL(cvt_weights_kernel, dim3(blocks), dim3(256), 0, stream,
                           W_hh, W_dec, Whh_b, Wdec_b);
    }

    // 2) h0 + zero adaptation state
    hipLaunchKernelGGL(init_kernel, dim3((Bc * NGc) / 256), dim3(256), 0, stream,
                       p0, W_init, hbuf0, va, zp);

    // 3) 100 dependent timesteps; ping-pong h buffers
    __bf16* hbufs[2] = {hbuf0, hbuf1};
    for (int t = 0; t < Tc; ++t) {
        hipLaunchKernelGGL(step_kernel, dim3(NGc / 64), dim3(32), 0, stream,
                           hbufs[t & 1], hbufs[(t + 1) & 1],
                           Whh_b, W_ih, v, va, zp, gbuf, t);
    }

    // 4) decoder GEMM -> fp32 output
    hipLaunchKernelGGL(decoder_kernel, dim3((Bc * Tc) / 64, NPc / 64), dim3(32), 0, stream,
                       gbuf, Wdec_b, out);
}
